// DosePredictionLoss_58342835749377
// MI455X (gfx1250) — matrix operand order repaired and checked by reading, not compiled, not run on hardware
//
#include <hip/hip_runtime.h>
#include <hip/hip_bf16.h>

typedef __attribute__((ext_vector_type(16))) _Float16 v16h;
typedef __attribute__((ext_vector_type(8)))  float    v8f;

#define QBINS 8192
#define NUM_BINS 60
#define MAX_DOSE 80.0f

// workspace layout (in floats / u32 words):
//   [0..4]            : sum_mse, sum_mse_ptv, sum_ptv, sum_mse_oar, sum_oar
//   [5..14]           : counts[10]
//   [16..1040)        : dvhP[16][64]
//   [1040..2064)      : dvhT[16][64]
//   [2064..2064+81920): histP[10][QBINS]  (u32)
//   [...+81920)       : histT[10][QBINS]  (u32)
#define WS_DVHP   16
#define WS_DVHT   (16 + 1024)
#define WS_HISTP  2064
#define WS_HISTT  (2064 + 10 * QBINS)
#define WS_WORDS  (2064 + 2 * 10 * QBINS)

// ---------------- Kernel 1: streaming reductions + dose histograms ----------------
__global__ void dose_main_reduce(const float4* __restrict__ o4,
                                 const float4* __restrict__ t4,
                                 const float4* __restrict__ m4,
                                 float* __restrict__ ws_f,
                                 unsigned* __restrict__ histP,
                                 unsigned* __restrict__ histT,
                                 int N4) {
    __shared__ float sred[16];
    if (threadIdx.x < 16) sred[threadIdx.x] = 0.0f;
    __syncthreads();

    float s0 = 0.f, s1 = 0.f, s2 = 0.f, s3 = 0.f, s4 = 0.f;
    float cnt[10];
#pragma unroll
    for (int s = 0; s < 10; ++s) cnt[s] = 0.f;

    const float qscale = (float)QBINS / MAX_DOSE;
    const int stride = gridDim.x * blockDim.x;
    for (int i = blockIdx.x * blockDim.x + threadIdx.x; i < N4; i += stride) {
        float4 o = o4[i];
        float4 t = t4[i];
        float4 mm[10];
#pragma unroll
        for (int s = 0; s < 10; ++s) mm[s] = m4[(size_t)s * N4 + i];

#pragma unroll
        for (int c = 0; c < 4; ++c) {
            float oc = ((const float*)&o)[c];
            float tc = ((const float*)&t)[c];
            float d = oc - tc;
            float mse = d * d;
            float msk[10];
#pragma unroll
            for (int s = 0; s < 10; ++s) msk[s] = ((const float*)&mm[s])[c];

            float ptv = fmaxf(fmaxf(msk[0], msk[1]), msk[2]);
            float oar = msk[3];
#pragma unroll
            for (int s = 4; s < 10; ++s) oar = fmaxf(oar, msk[s]);
            float oon = oar * (1.0f - ptv);

            s0 += mse;
            s1 += mse * ptv;
            s2 += ptv;
            s3 += mse * oon;
            s4 += oon;

            int qo = (int)(oc * qscale);
            int qt = (int)(tc * qscale);
            qo = qo < 0 ? 0 : (qo > QBINS - 1 ? QBINS - 1 : qo);
            qt = qt < 0 ? 0 : (qt > QBINS - 1 ? QBINS - 1 : qt);
#pragma unroll
            for (int s = 0; s < 10; ++s) {
                cnt[s] += msk[s];
                if (msk[s] > 0.5f) {
                    atomicAdd(histP + s * QBINS + qo, 1u);
                    atomicAdd(histT + s * QBINS + qt, 1u);
                }
            }
        }
    }

    atomicAdd(&sred[0], s0);
    atomicAdd(&sred[1], s1);
    atomicAdd(&sred[2], s2);
    atomicAdd(&sred[3], s3);
    atomicAdd(&sred[4], s4);
#pragma unroll
    for (int s = 0; s < 10; ++s) atomicAdd(&sred[5 + s], cnt[s]);
    __syncthreads();
    if (threadIdx.x < 15) atomicAdd(&ws_f[threadIdx.x], sred[threadIdx.x]);
}

// ---------------- Kernel 2: DVH = hist[10,Q] x sigma[Q,64] via WMMA f16 ----------------
__global__ void dose_dvh_wmma(const unsigned* __restrict__ histP,
                              const unsigned* __restrict__ histT,
                              float* __restrict__ dvhP,
                              float* __restrict__ dvhT) {
    const int lane = threadIdx.x & 31;
    const int wid  = (blockIdx.x * blockDim.x + threadIdx.x) >> 5;
    const int nw   = (gridDim.x * blockDim.x) >> 5;
    const int m    = lane & 15;      // row (structure) for A, bin-in-group for B/C
    const int hi   = lane >> 4;      // half-wave select
    const int koff = hi ? 16 : 0;    // A-layout K offset for this half
    const int kb   = lane & 16;      // B-layout K base for this half
    const float dq = MAX_DOSE / (float)QBINS;
    const float rstep = __expf(-dq); // e_{q+1} = e_q * rstep (geometric recurrence)

    // C_b = exp(bin_b), one bin per lane per group (bins padded 60->64, clamp)
    float cb[4];
#pragma unroll
    for (int g = 0; g < 4; ++g) {
        int b = g * 16 + m;
        if (b > NUM_BINS - 1) b = NUM_BINS - 1;
        cb[g] = __expf((float)b * (MAX_DOSE / (float)(NUM_BINS - 1)));
    }

    v8f accP[4] = {};
    v8f accT[4] = {};

    const int nchunks = QBINS / 32;
    for (int c = wid; c < nchunks; c += nw) {   // wave-uniform loop: EXEC all-ones at WMMA
        const int qb = c * 32;

        // e_h = exp(-dose_{qb+kb+h}) via one exp + geometric recurrence (no shfl/LDS)
        float eh[16];
        eh[0] = __expf(-((float)(qb + kb) + 0.5f) * dq);
#pragma unroll
        for (int h = 1; h < 16; ++h) eh[h] = eh[h - 1] * rstep;

        // A matrices (16-bit 16x32 layout): lane holds row m; halves 0..15 = K koff..koff+15.
        // 16 consecutive u32 per half -> uint4 loads.
        v16h aP, aT;
        {
            uint4 z = make_uint4(0u, 0u, 0u, 0u);
            const uint4* hp = (const uint4*)(histP + m * QBINS + qb + koff);
            const uint4* ht = (const uint4*)(histT + m * QBINS + qb + koff);
            uint4 p0 = (m < 10) ? hp[0] : z;
            uint4 p1 = (m < 10) ? hp[1] : z;
            uint4 p2 = (m < 10) ? hp[2] : z;
            uint4 p3 = (m < 10) ? hp[3] : z;
            uint4 t0 = (m < 10) ? ht[0] : z;
            uint4 t1 = (m < 10) ? ht[1] : z;
            uint4 t2 = (m < 10) ? ht[2] : z;
            uint4 t3 = (m < 10) ? ht[3] : z;
            const unsigned* pw[4] = { (const unsigned*)&p0, (const unsigned*)&p1,
                                      (const unsigned*)&p2, (const unsigned*)&p3 };
            const unsigned* tw[4] = { (const unsigned*)&t0, (const unsigned*)&t1,
                                      (const unsigned*)&t2, (const unsigned*)&t3 };
#pragma unroll
            for (int v = 0; v < 4; ++v) {
#pragma unroll
                for (int j = 0; j < 4; ++j) {
                    aP[v * 4 + j] = (_Float16)(float)pw[v][j];
                    aT[v * 4 + j] = (_Float16)(float)tw[v][j];
                }
            }
        }

#pragma unroll
        for (int g = 0; g < 4; ++g) {
            // B (16-bit 32x16 layout): lane = column n, half h = K kb+h
            v16h bM;
#pragma unroll
            for (int h = 0; h < 16; ++h) {
                float sg = __builtin_amdgcn_rcpf(1.0f + eh[h] * cb[g]);
                bM[h] = (_Float16)sg;
            }
            accP[g] = __builtin_amdgcn_wmma_f32_16x16x32_f16(
                false, aP, false, bM, (short)0, accP[g], false, false);
            accT[g] = __builtin_amdgcn_wmma_f32_16x16x32_f16(
                false, aT, false, bM, (short)0, accT[g], false, false);
        }
    }

    // C/D layout: lanes 0-15 -> M=r, N=lane; lanes 16-31 -> M=r+8, N=lane-16
#pragma unroll
    for (int g = 0; g < 4; ++g) {
#pragma unroll
        for (int r = 0; r < 8; ++r) {
            int row = r + hi * 8;
            int bin = g * 16 + m;
            atomicAdd(&dvhP[row * 64 + bin], accP[g][r]);
            atomicAdd(&dvhT[row * 64 + bin], accT[g][r]);
        }
    }
}

// ---------------- Kernel 3: finalize scalar loss ----------------
__global__ void dose_finalize(const float* __restrict__ ws_f,
                              float* __restrict__ out, float invN) {
    if (threadIdx.x == 0 && blockIdx.x == 0) {
        const float* dvhP = ws_f + WS_DVHP;
        const float* dvhT = ws_f + WS_DVHT;
        float Lg   = ws_f[0] * invN;
        float Lptv = ws_f[1] * 3.0f / (ws_f[2] + 1e-6f);
        float Loar = ws_f[3] * 1.5f / (ws_f[4] + 1e-6f);
        float sum = 0.0f;
        for (int s = 0; s < 10; ++s) {
            float c  = ws_f[5 + s];
            float cs = fmaxf(c, 1.0f);
            float a = 0.0f;
            for (int b = 0; b < NUM_BINS; ++b)
                a += fabsf(dvhP[s * 64 + b] - dvhT[s * 64 + b]);
            float ls = (a / cs) / (float)NUM_BINS;
            if (c >= 1.0f) sum += ls;
        }
        float Ldvh = (sum / 10.0f) * 0.5f;
        out[0] = Lg + Lptv + Loar + Ldvh;
    }
}

extern "C" void kernel_launch(void* const* d_in, const int* in_sizes, int n_in,
                              void* d_out, int out_size, void* d_ws, size_t ws_size,
                              hipStream_t stream) {
    const float4* o4 = (const float4*)d_in[0];
    const float4* t4 = (const float4*)d_in[1];
    const float4* m4 = (const float4*)d_in[2];
    float* out = (float*)d_out;

    const int N  = in_sizes[0];          // 128^3
    const int N4 = N / 4;

    float*    ws_f  = (float*)d_ws;
    unsigned* ws_u  = (unsigned*)d_ws;
    unsigned* histP = ws_u + WS_HISTP;
    unsigned* histT = ws_u + WS_HISTT;
    float*    dvhP  = ws_f + WS_DVHP;
    float*    dvhT  = ws_f + WS_DVHT;

    // zero accumulators + histograms every call (graph-capture safe, deterministic)
    hipMemsetAsync(d_ws, 0, (size_t)WS_WORDS * sizeof(float), stream);

    dose_main_reduce<<<1024, 256, 0, stream>>>(o4, t4, m4, ws_f, histP, histT, N4);

    // QBINS/32 = 256 K-chunks; 32 blocks x 256 threads = 256 waves, one chunk each
    dose_dvh_wmma<<<32, 256, 0, stream>>>(histP, histT, dvhP, dvhT);

    dose_finalize<<<1, 32, 0, stream>>>(ws_f, out, 1.0f / (float)N);
}
